// TK_v4_pos_42674795053883
// MI455X (gfx1250) — matrix operand order repaired and checked by reading, not compiled, not run on hardware
//
#include <hip/hip_runtime.h>
#include <stdint.h>

#define B_  16
#define Q_  32
#define DL_ 1024
#define E_  256
#define H_  8
#define L_  2
#define F_  1024
#define NK_ 11
#define DH_ 32

typedef __bf16 bf16_t;
typedef bf16_t v16bf __attribute__((ext_vector_type(16)));
typedef bf16_t v2bf  __attribute__((ext_vector_type(2)));
typedef float  v8f   __attribute__((ext_vector_type(8)));
typedef uint32_t u32;
typedef uint16_t u16;

union Frag {
  v16bf v;
  uint4 u4[2];
  u32   u[8];
  u16   h[16];
};

// f32 -> bf16 via compiler's native bf16 (lowers to hw cvt where available).
__device__ __forceinline__ u16 f2bf(float f) {
  union { bf16_t h; u16 u; } r;
  r.h = (bf16_t)f;
  return r.u;
}
__device__ __forceinline__ u32 pack2(float a, float b) {
  union { v2bf v; u32 u; } r;
  r.v[0] = (bf16_t)a;
  r.v[1] = (bf16_t)b;
  return r.u;
}
__device__ __forceinline__ v8f vzero() {
  v8f z;
#pragma unroll
  for (int i = 0; i < 8; ++i) z[i] = 0.0f;
  return z;
}
__device__ __forceinline__ v8f wmma_bf16(const Frag& a, const Frag& b, v8f c) {
  return __builtin_amdgcn_wmma_f32_16x16x32_bf16(false, a.v, false, b.v,
                                                 (short)0, c, false, false);
}

// A-fragment (16x32, MxK) from f32 row-major, converting to bf16.
// lane l: m = l&15; k-halves [k0,k0+8) and [16+k0,16+k0+8), k0 = (l<16)?0:8
__device__ __forceinline__ Frag load_a_f32(const float* X, int ld) {
  int l = (int)(threadIdx.x & 31);
  const float* p = X + (size_t)(l & 15) * ld + ((l < 16) ? 0 : 8);
  float4 a0 = *(const float4*)(p);
  float4 a1 = *(const float4*)(p + 4);
  float4 a2 = *(const float4*)(p + 16);
  float4 a3 = *(const float4*)(p + 20);
  Frag f;
  f.u[0] = pack2(a0.x, a0.y); f.u[1] = pack2(a0.z, a0.w);
  f.u[2] = pack2(a1.x, a1.y); f.u[3] = pack2(a1.z, a1.w);
  f.u[4] = pack2(a2.x, a2.y); f.u[5] = pack2(a2.z, a2.w);
  f.u[6] = pack2(a3.x, a3.y); f.u[7] = pack2(a3.z, a3.w);
  return f;
}
// A-fragment from bf16 row-major (contiguous 16B chunks per lane).
__device__ __forceinline__ Frag load_a_bf16(const u16* X, int ld) {
  int l = (int)(threadIdx.x & 31);
  const u16* p = X + (size_t)(l & 15) * ld + ((l < 16) ? 0 : 8);
  Frag f;
  f.u4[0] = *(const uint4*)(p);
  f.u4[1] = *(const uint4*)(p + 16);
  return f;
}
// B-fragment (32x16, KxN) from bf16 (K x N) row-major: lane holds column
// n = l&15, k-halves [k0,k0+8) and [16+k0,16+k0+8), k0 = (l<16)?0:8
__device__ __forceinline__ Frag load_b_bf16(const u16* Wt, int ld) {
  int l = (int)(threadIdx.x & 31);
  const u16* p = Wt + (l & 15);
  int k0 = (l < 16) ? 0 : 8;
  Frag f;
#pragma unroll
  for (int s = 0; s < 8; ++s) {
    f.h[s] = p[(size_t)(k0 + s) * ld];
    f.h[8 + s] = p[(size_t)(16 + k0 + s) * ld];
  }
  return f;
}

__device__ __forceinline__ float halfmax(float v) {
#pragma unroll
  for (int s = 1; s <= 8; s <<= 1) v = fmaxf(v, __shfl_xor(v, s, 32));
  return v;
}
__device__ __forceinline__ float halfsum(float v) {
#pragma unroll
  for (int s = 1; s <= 8; s <<= 1) v += __shfl_xor(v, s, 32);
  return v;
}
__device__ __forceinline__ float wavesum(float v) {
#pragma unroll
  for (int s = 1; s <= 16; s <<= 1) v += __shfl_xor(v, s, 32);
  return v;
}

// ---------------- prep kernels ----------------

// src (R,C) f32 row-major -> dst (C,R) bf16 row-major
__global__ void k_transpose_bf16(const float* __restrict__ src, u16* __restrict__ dst,
                                 int R, int C) {
  long i = (long)blockIdx.x * blockDim.x + threadIdx.x;
  if (i >= (long)R * C) return;
  int r = (int)(i / C), c = (int)(i % C);
  dst[(size_t)c * R + r] = f2bf(src[i]);
}

// X = (use_emb ? emb*mask : 0) + pos (pos broadcast over batch)
__global__ void k_embed(const float* __restrict__ emb, const float* __restrict__ mask,
                        const float* __restrict__ pos, float* __restrict__ X,
                        int S, int use_emb) {
  long i = (long)blockIdx.x * blockDim.x + threadIdx.x;
  if (i >= (long)B_ * S * E_) return;
  int e = (int)(i % E_);
  long bs = i / E_;
  int s = (int)(bs % S);
  float v = pos[(size_t)s * E_ + e];
  if (use_emb) v += emb[i] * mask[bs];
  X[i] = v;
}

// QKV (B*S,3E) f32 -> Qb (B,H,S,dh) bf16, KbT (B,H,dh,S) bf16, Vb (B,H,S,dh) bf16
__global__ void k_pack_qkv(const float* __restrict__ QKV, u16* __restrict__ Qb,
                           u16* __restrict__ KbT, u16* __restrict__ Vb, int S) {
  long i = (long)blockIdx.x * blockDim.x + threadIdx.x;
  if (i >= (long)B_ * S * E_) return;
  int e = (int)(i % E_);
  long bs = i / E_;
  int s = (int)(bs % S);
  int b = (int)(bs / S);
  int h = e >> 5, d = e & 31;
  size_t bh = (size_t)b * H_ + h;
  size_t qoff = (size_t)bs * (3 * E_);
  Qb[(bh * S + s) * DH_ + d] = f2bf(QKV[qoff + e]);
  KbT[(bh * DH_ + d) * S + s] = f2bf(QKV[qoff + E_ + e]);
  Vb[(bh * S + s) * DH_ + d] = f2bf(QKV[qoff + 2 * E_ + e]);
}

// ---------------- GEMM: Y = relu?(X @ Wt + bias) ----------------
// X (M,Kd) f32, Wt (Kd,N) bf16, Y (M,N) f32. One wave -> 32x64 strip
// (two A fragments share four B fragments -> 8 WMMAs per K-step).
__global__ void k_gemm(const float* __restrict__ X, const u16* __restrict__ Wt,
                       const float* __restrict__ bias, float* __restrict__ Y,
                       int M, int N, int Kd, int relu) {
  int wid = (int)(((long)blockIdx.x * blockDim.x + threadIdx.x) >> 5);
  int ntiles = N >> 6;
  int tw = (M >> 5) * ntiles;
  if (wid >= tw) return;
  int rt = wid / ntiles, ct = wid % ntiles;
  int l = (int)(threadIdx.x & 31);
  v8f acc[8];
#pragma unroll
  for (int i = 0; i < 8; ++i) acc[i] = vzero();
  const float* Xt0 = X + (size_t)rt * 32 * Kd;
  const float* Xt1 = Xt0 + (size_t)16 * Kd;
  const u16* Wb = Wt + ct * 64;
  int krow = (l & 15) + ((l < 16) ? 0 : 16);
  for (int kk = 0; kk < Kd; kk += 32) {
    // prefetch next K-step's weight rows (global_prefetch_b8)
    __builtin_prefetch(Wb + (size_t)(kk + 32 + krow) * N, 0, 1);
    Frag a0 = load_a_f32(Xt0 + kk, Kd);
    Frag a1 = load_a_f32(Xt1 + kk, Kd);
#pragma unroll
    for (int t = 0; t < 4; ++t) {
      Frag b = load_b_bf16(Wb + (size_t)kk * N + t * 16, N);
      acc[t]     = wmma_bf16(a0, b, acc[t]);
      acc[4 + t] = wmma_bf16(a1, b, acc[4 + t]);
    }
  }
  int row0 = rt * 32 + ((l < 16) ? 0 : 8);
  int col0 = ct * 64 + (l & 15);
#pragma unroll
  for (int r = 0; r < 2; ++r) {
#pragma unroll
    for (int t = 0; t < 4; ++t) {
      v8f a = acc[r * 4 + t];
      int col = col0 + t * 16;
      float bz = bias[col];
#pragma unroll
      for (int j = 0; j < 8; ++j) {
        float v = a[j] + bz;
        if (relu) v = fmaxf(v, 0.0f);
        Y[(size_t)(row0 + r * 16 + j) * N + col] = v;
      }
    }
  }
}

// ---------------- flash attention ----------------
// One wave per (b,h,16-query tile). dh=32. Online softmax, P relayout via LDS.
__global__ void k_attn(const u16* __restrict__ Qb, const u16* __restrict__ KbT,
                       const u16* __restrict__ Vb, const float* __restrict__ mask,
                       float* __restrict__ AO, int S) {
  __shared__ __align__(16) u16 plds[8][16 * 32];
  int wib = (int)(threadIdx.x >> 5);
  int wid = (int)(blockIdx.x * (blockDim.x >> 5) + wib);
  int qtiles = S >> 4;
  if (wid >= B_ * H_ * qtiles) return;
  int qt = wid % qtiles;
  int bh = wid / qtiles;
  int b = bh / H_, h = bh % H_;
  int l = (int)(threadIdx.x & 31);
  const u16* Qbase = Qb + (size_t)bh * S * DH_;
  const u16* Kbase = KbT + (size_t)bh * DH_ * S;
  const u16* Vbase = Vb + (size_t)bh * S * DH_;
  Frag qf = load_a_bf16(Qbase + (size_t)qt * 16 * DH_, DH_);
  float Mr[8], Lr[8];
  v8f o0 = vzero(), o1 = vzero();
#pragma unroll
  for (int j = 0; j < 8; ++j) { Mr[j] = -1e30f; Lr[j] = 0.0f; }
  const float scale = 0.17677669529663687f;  // 1/sqrt(32)
  int krow = (l & 15) + ((l < 16) ? 0 : 16);
  for (int kb = 0; kb < S; kb += 32) {
    __builtin_prefetch(Kbase + (size_t)krow * S + kb + 32, 0, 1);
    __builtin_prefetch(Vbase + (size_t)(kb + 32 + krow) * DH_, 0, 1);
    Frag kf0 = load_b_bf16(Kbase + kb, S);
    Frag kf1 = load_b_bf16(Kbase + kb + 16, S);
    v8f s0 = wmma_bf16(qf, kf0, vzero());
    v8f s1 = wmma_bf16(qf, kf1, vzero());
    float kb0 = (mask[b * S + kb + (l & 15)] > 0.5f) ? 0.0f : -1e9f;
    float kb1 = (mask[b * S + kb + 16 + (l & 15)] > 0.5f) ? 0.0f : -1e9f;
    float p0[8], p1[8];
#pragma unroll
    for (int j = 0; j < 8; ++j) {
      float a0 = s0[j] * scale + kb0;
      float a1 = s1[j] * scale + kb1;
      float nm = fmaxf(Mr[j], halfmax(fmaxf(a0, a1)));
      float fsc = __expf(Mr[j] - nm);
      Mr[j] = nm;
      Lr[j] *= fsc; o0[j] *= fsc; o1[j] *= fsc;
      float e0 = __expf(a0 - nm);
      float e1 = __expf(a1 - nm);
      Lr[j] += halfsum(e0 + e1);
      p0[j] = e0; p1[j] = e1;
    }
    {   // D-layout -> LDS (bf16), 16x32 tile
      int rb = (l < 16) ? 0 : 8;
      int c = l & 15;
      u16* P = &plds[wib][0];
#pragma unroll
      for (int j = 0; j < 8; ++j) {
        P[(rb + j) * 32 + c] = f2bf(p0[j]);
        P[(rb + j) * 32 + 16 + c] = f2bf(p1[j]);
      }
    }
    asm volatile("s_wait_dscnt 0x0" ::: "memory");
    Frag pf = load_a_bf16(&plds[wib][0], 32);
    Frag vf0 = load_b_bf16(Vbase + (size_t)kb * DH_, DH_);
    Frag vf1 = load_b_bf16(Vbase + (size_t)kb * DH_ + 16, DH_);
    o0 = wmma_bf16(pf, vf0, o0);
    o1 = wmma_bf16(pf, vf1, o1);
  }
  int rb = qt * 16 + ((l < 16) ? 0 : 8);
  int c = l & 15;
#pragma unroll
  for (int j = 0; j < 8; ++j) {
    float inv = 1.0f / fmaxf(Lr[j], 1e-20f);
    size_t row = (size_t)b * S + rb + j;
    AO[row * E_ + h * DH_ + c] = o0[j] * inv;
    AO[row * E_ + h * DH_ + 16 + c] = o1[j] * inv;
  }
}

// ---------------- residual + layernorm (in place on X) ----------------
__global__ void k_add_ln(float* __restrict__ X, const float* __restrict__ R,
                         const float* __restrict__ sc, const float* __restrict__ bi,
                         int M) {
  int wid = (int)(((long)blockIdx.x * blockDim.x + threadIdx.x) >> 5);
  if (wid >= M) return;
  int l = (int)(threadIdx.x & 31);
  size_t base = (size_t)wid * E_;
  float v[8]; float s = 0.0f;
#pragma unroll
  for (int i = 0; i < 8; ++i) {
    int c = l + 32 * i;
    v[i] = X[base + c] + R[base + c];
    s += v[i];
  }
  float mean = wavesum(s) * (1.0f / E_);
  float var = 0.0f;
#pragma unroll
  for (int i = 0; i < 8; ++i) { float d = v[i] - mean; var += d * d; }
  var = wavesum(var) * (1.0f / E_);
  float inv = rsqrtf(var + 1e-5f);
#pragma unroll
  for (int i = 0; i < 8; ++i) {
    int c = l + 32 * i;
    X[base + c] = (v[i] - mean) * inv * sc[c] + bi[c];
  }
}

// ---------------- subtract + L2 normalize ----------------
__global__ void k_subnorm_q(const float* __restrict__ A, const float* __restrict__ Bm,
                            u16* __restrict__ out, int M) {
  int wid = (int)(((long)blockIdx.x * blockDim.x + threadIdx.x) >> 5);
  if (wid >= M) return;
  int l = (int)(threadIdx.x & 31);
  size_t base = (size_t)wid * E_;
  float d[8]; float s = 0.0f;
#pragma unroll
  for (int i = 0; i < 8; ++i) {
    int c = l + 32 * i;
    d[i] = A[base + c] - Bm[base + c];
    s += d[i] * d[i];
  }
  s = wavesum(s);
  float inv = 1.0f / (sqrtf(s) + 1e-13f);
#pragma unroll
  for (int i = 0; i < 8; ++i) out[base + l + 32 * i] = f2bf(d[i] * inv);
}

__global__ void k_subnorm_d(const float* __restrict__ A, const float* __restrict__ Bm,
                            u16* __restrict__ outT) {
  int wid = (int)(((long)blockIdx.x * blockDim.x + threadIdx.x) >> 5);
  if (wid >= B_ * DL_) return;
  int b = wid / DL_, dd = wid % DL_;
  int l = (int)(threadIdx.x & 31);
  size_t base = (size_t)wid * E_;
  float d[8]; float s = 0.0f;
#pragma unroll
  for (int i = 0; i < 8; ++i) {
    int c = l + 32 * i;
    d[i] = A[base + c] - Bm[base + c];
    s += d[i] * d[i];
  }
  s = wavesum(s);
  float inv = 1.0f / (sqrtf(s) + 1e-13f);
#pragma unroll
  for (int i = 0; i < 8; ++i) {
    int c = l + 32 * i;
    outT[((size_t)b * E_ + c) * DL_ + dd] = f2bf(d[i] * inv);
  }
}

__global__ void k_zero(float* __restrict__ p, int n) {
  int i = (int)((long)blockIdx.x * blockDim.x + threadIdx.x);
  if (i < n) p[i] = 0.0f;
}

// ---------------- cosine matrix (WMMA) + Gaussian kernel pooling ----------------
// One wave per (b, q-tile of 16, 128-doc chunk).
__global__ void k_cospool(const u16* __restrict__ qn, const u16* __restrict__ dnT,
                          const float* __restrict__ dmask, const float* __restrict__ mu,
                          const float* __restrict__ sigma, float* __restrict__ pkq) {
  int wid = (int)(((long)blockIdx.x * blockDim.x + threadIdx.x) >> 5);
  if (wid >= B_ * 2 * 8) return;
  int dchunk = wid & 7;
  int qt = (wid >> 3) & 1;
  int b = wid >> 4;
  int l = (int)(threadIdx.x & 31);
  float muv[NK_], c2[NK_];
#pragma unroll
  for (int k = 0; k < NK_; ++k) {
    muv[k] = mu[k];
    float sg = sigma[k];
    c2[k] = 0.5f / (sg * sg);
  }
  Frag af[8];
  const u16* qbase = qn + ((size_t)b * Q_ + qt * 16) * E_;
#pragma unroll
  for (int t = 0; t < 8; ++t) af[t] = load_a_bf16(qbase + t * 32, E_);
  float acc[8][NK_];
#pragma unroll
  for (int j = 0; j < 8; ++j)
#pragma unroll
    for (int k = 0; k < NK_; ++k) acc[j][k] = 0.0f;
  for (int dt = 0; dt < 8; ++dt) {
    int d0 = dchunk * 128 + dt * 16;
    v8f cacc = vzero();
#pragma unroll
    for (int t = 0; t < 8; ++t) {
      Frag bb = load_b_bf16(dnT + ((size_t)b * E_ + t * 32) * DL_ + d0, DL_);
      cacc = wmma_bf16(af[t], bb, cacc);
    }
    float dm = dmask[b * DL_ + d0 + (l & 15)];
    if (dm > 0.5f) {
#pragma unroll
      for (int j = 0; j < 8; ++j) {
        float c = cacc[j];
#pragma unroll
        for (int k = 0; k < NK_; ++k) {
          float df = c - muv[k];
          acc[j][k] += __expf(-df * df * c2[k]);
        }
      }
    }
  }
  int rb = (l < 16) ? 0 : 8;
#pragma unroll
  for (int j = 0; j < 8; ++j) {
#pragma unroll
    for (int k = 0; k < NK_; ++k) {
      float v = halfsum(acc[j][k]);
      if ((l & 15) == 0)
        atomicAdd(&pkq[((size_t)b * Q_ + qt * 16 + rb + j) * NK_ + k], v);
    }
  }
}

// ---------------- final: log pooling + dense head ----------------
__global__ void k_final(const float* __restrict__ pkq, const float* __restrict__ qmask,
                        const float* __restrict__ dense, const float* __restrict__ nnsc,
                        float* __restrict__ out) {
  __shared__ float part[B_][NK_ + 1];
  int t = (int)threadIdx.x;
  int b = t >> 4, k = t & 15;
  if (b < B_ && k < NK_) {
    float ns = nnsc[0];
    float s = 0.0f;
    for (int q = 0; q < Q_; ++q) {
      float v = fmaxf(pkq[((size_t)b * Q_ + q) * NK_ + k], 1e-10f);
      s += log2f(v) * ns * qmask[b * Q_ + q];
    }
    part[b][k] = s;
  }
  __syncthreads();
  if (t < B_) {
    float s = 0.0f;
    for (int k2 = 0; k2 < NK_; ++k2) s += part[t][k2] * dense[k2];
    out[t] = s;
  }
}

// ======================= host side =======================

static inline size_t align256(size_t x) { return (x + 255) & ~(size_t)255; }

extern "C" void kernel_launch(void* const* d_in, const int* in_sizes, int n_in,
                              void* d_out, int out_size, void* d_ws, size_t ws_size,
                              hipStream_t stream) {
  (void)in_sizes; (void)n_in; (void)out_size; (void)ws_size;
  const float* qe    = (const float*)d_in[0];
  const float* de    = (const float*)d_in[1];
  const float* qmask = (const float*)d_in[2];
  const float* dmask = (const float*)d_in[3];
  const float* Wqkv  = (const float*)d_in[4];
  const float* bqkv  = (const float*)d_in[5];
  const float* Wo    = (const float*)d_in[6];
  const float* bo    = (const float*)d_in[7];
  const float* W1    = (const float*)d_in[8];
  const float* b1    = (const float*)d_in[9];
  const float* W2    = (const float*)d_in[10];
  const float* b2    = (const float*)d_in[11];
  const float* ln1s  = (const float*)d_in[12];
  const float* ln1b  = (const float*)d_in[13];
  const float* ln2s  = (const float*)d_in[14];
  const float* ln2b  = (const float*)d_in[15];
  const float* dense = (const float*)d_in[16];
  const float* nnsc  = (const float*)d_in[19];
  const float* pos   = (const float*)d_in[20];
  const float* mu    = (const float*)d_in[21];
  const float* sigma = (const float*)d_in[22];
  float* out = (float*)d_out;

  char* ws = (char*)d_ws;
  size_t off = 0;
  auto take = [&](size_t bytes) { char* p = ws + off; off = align256(off + bytes); return p; };

  const int Mdoc = B_ * DL_;
  u16* WqkvT = (u16*)take((size_t)L_ * E_ * 3 * E_ * 2);
  u16* WoT   = (u16*)take((size_t)L_ * E_ * E_ * 2);
  u16* W1T   = (u16*)take((size_t)L_ * E_ * F_ * 2);
  u16* W2T   = (u16*)take((size_t)L_ * F_ * E_ * 2);
  float* X    = (float*)take((size_t)Mdoc * E_ * 4);
  float* QKV  = (float*)take((size_t)Mdoc * 3 * E_ * 4);   // 48MB
  u16*   Qb   = (u16*)take((size_t)Mdoc * E_ * 2);         // 8MB
  u16*   KbT  = (u16*)take((size_t)Mdoc * E_ * 2);         // 8MB
  u16*   Vb   = (u16*)take((size_t)Mdoc * E_ * 2);
  float* AO   = (float*)take((size_t)Mdoc * E_ * 4);
  float* Obuf = (float*)take((size_t)Mdoc * E_ * 4);
  float* d1   = (float*)take((size_t)Mdoc * E_ * 4);
  u16*   dnT  = (u16*)take((size_t)B_ * E_ * DL_ * 2);
  u16*   qn   = (u16*)take((size_t)B_ * Q_ * E_ * 2);
  float* pkq  = (float*)take((size_t)B_ * Q_ * NK_ * 4);
  // FF1 (64MB) aliases QKV+Qb+KbT (all dead by FFN time); FF2 aliases AO.
  float* FF1 = QKV;
  float* FF2 = AO;

  auto cdiv = [](long a, long b) { return (int)((a + b - 1) / b); };

  // --- weight prep (transpose + bf16) ---
  for (int l = 0; l < L_; ++l) {
    k_transpose_bf16<<<cdiv((long)3 * E_ * E_, 256), 256, 0, stream>>>(
        Wqkv + (size_t)l * 3 * E_ * E_, WqkvT + (size_t)l * E_ * 3 * E_, 3 * E_, E_);
    k_transpose_bf16<<<cdiv((long)E_ * E_, 256), 256, 0, stream>>>(
        Wo + (size_t)l * E_ * E_, WoT + (size_t)l * E_ * E_, E_, E_);
    k_transpose_bf16<<<cdiv((long)F_ * E_, 256), 256, 0, stream>>>(
        W1 + (size_t)l * F_ * E_, W1T + (size_t)l * E_ * F_, F_, E_);
    k_transpose_bf16<<<cdiv((long)E_ * F_, 256), 256, 0, stream>>>(
        W2 + (size_t)l * E_ * F_, W2T + (size_t)l * F_ * E_, E_, F_);
  }

  auto gemm = [&](const float* Xp, const u16* Wt, const float* bias, float* Y,
                  int M, int N, int Kd, int relu) {
    long waves = (long)(M / 32) * (N / 64);
    k_gemm<<<cdiv(waves * 32, 256), 256, 0, stream>>>(Xp, Wt, bias, Y, M, N, Kd, relu);
  };

  auto run_tr = [&](float* Xbuf, const float* mk, int S) {
    int M = B_ * S;
    for (int l = 0; l < L_; ++l) {
      gemm(Xbuf, WqkvT + (size_t)l * E_ * 3 * E_, bqkv + l * 3 * E_, QKV, M, 3 * E_, E_, 0);
      k_pack_qkv<<<cdiv((long)M * E_, 256), 256, 0, stream>>>(QKV, Qb, KbT, Vb, S);
      long awaves = (long)B_ * H_ * (S / 16);
      k_attn<<<cdiv(awaves * 32, 256), 256, 0, stream>>>(Qb, KbT, Vb, mk, AO, S);
      gemm(AO, WoT + (size_t)l * E_ * E_, bo + l * E_, Obuf, M, E_, E_, 0);
      k_add_ln<<<cdiv((long)M * 32, 256), 256, 0, stream>>>(Xbuf, Obuf, ln1s + l * E_, ln1b + l * E_, M);
      gemm(Xbuf, W1T + (size_t)l * E_ * F_, b1 + l * F_, FF1, M, F_, E_, 1);
      gemm(FF1, W2T + (size_t)l * F_ * E_, b2 + l * E_, FF2, M, E_, F_, 0);
      k_add_ln<<<cdiv((long)M * 32, 256), 256, 0, stream>>>(Xbuf, FF2, ln2s + l * E_, ln2b + l * E_, M);
    }
  };

  // --- document streams ---
  k_embed<<<cdiv((long)Mdoc * E_, 256), 256, 0, stream>>>(de, dmask, pos, X, DL_, 1);
  run_tr(X, dmask, DL_);
  hipMemcpyAsync(d1, X, (size_t)Mdoc * E_ * 4, hipMemcpyDeviceToDevice, stream);
  k_embed<<<cdiv((long)Mdoc * E_, 256), 256, 0, stream>>>(de, dmask, pos, X, DL_, 0);
  run_tr(X, dmask, DL_);
  k_subnorm_d<<<cdiv((long)Mdoc * 32, 256), 256, 0, stream>>>(d1, X, dnT);

  // --- query streams (reuse buffers; d1 reused for q_ctx1) ---
  int Mq = B_ * Q_;
  k_embed<<<cdiv((long)Mq * E_, 256), 256, 0, stream>>>(qe, qmask, pos, X, Q_, 1);
  run_tr(X, qmask, Q_);
  hipMemcpyAsync(d1, X, (size_t)Mq * E_ * 4, hipMemcpyDeviceToDevice, stream);
  k_embed<<<cdiv((long)Mq * E_, 256), 256, 0, stream>>>(qe, qmask, pos, X, Q_, 0);
  run_tr(X, qmask, Q_);
  k_subnorm_q<<<cdiv((long)Mq * 32, 256), 256, 0, stream>>>(d1, X, qn, Mq);

  // --- scoring ---
  k_zero<<<cdiv((long)B_ * Q_ * NK_, 256), 256, 0, stream>>>(pkq, B_ * Q_ * NK_);
  k_cospool<<<cdiv((long)B_ * 2 * 8 * 32, 256), 256, 0, stream>>>(qn, dnT, dmask, mu, sigma, pkq);
  k_final<<<1, 256, 0, stream>>>(pkq, qmask, dense, nnsc, out);
}
